// EncoderBlock_90829968376275
// MI455X (gfx1250) — compile-verified, matrix operand order checked by reading, not run on hardware
//
#include <hip/hip_runtime.h>

// ---------------------------------------------------------------------------
// CDNA5 (gfx1250) encoder block.
//   * All matmuls: bf16 x bf16 -> f32 via v_wmma_f32_16x16x32_bf16 (wave32).
//   * GEMM B tiles are DMA'd to LDS with the Tensor Data Mover
//     (tensor_load_to_lds + s_wait_tensorcnt), double-buffered.
//   * Conv1d(SAME,3) = 3 accumulating GEMMs over a zero-padded activation.
//   * Attention: flash-style, WMMA for QK^T and P@V, online softmax in LDS.
//   * Elementwise / RMS / RoPE / SiLU: fp32 VALU kernels (bandwidth-trivial).
// ---------------------------------------------------------------------------

typedef __attribute__((ext_vector_type(16))) __bf16 v16bf;
typedef __attribute__((ext_vector_type(8)))  __bf16 v8bf;
typedef __attribute__((ext_vector_type(8)))  float  v8f;
typedef __attribute__((ext_vector_type(4)))  unsigned int u32x4_t;
typedef __attribute__((ext_vector_type(8)))  int i32x8_t;
typedef __attribute__((ext_vector_type(4)))  int i32x4_t;

union FragBf { v16bf v; v8bf h[2]; };

#define EPSF 1e-5f

// ISA 7.12.2: 16-bit A (16x32) fragment.  lane&15 = row, lane>>4 selects the
// K=8..15 / 24..31 half.  Two 16-byte loads per lane.  Same layout serves the
// B fragment when B^T (N x K, row-major) is supplied.
__device__ __forceinline__ v16bf load_frag(const __bf16* p0, int stride, int lane) {
    const __bf16* p = p0 + (size_t)(lane & 15) * (size_t)stride + ((lane >> 4) ? 8 : 0);
    FragBf f;
    f.h[0] = *(const v8bf*)(p);
    f.h[1] = *(const v8bf*)(p + 16);
    return f.v;
}

__device__ __forceinline__ v8f wmma_bf16(v16bf a, v16bf b, v8f c) {
    return __builtin_amdgcn_wmma_f32_16x16x32_bf16(false, a, false, b, (short)0, c,
                                                   false, false);
}

// ---------------------------------------------------------------------------
// TDM: 2-D tile (tile_rows x tile_cols, 2-byte elements) global -> LDS.
// Builds D# group0/group1 per CDNA5 ISA ch.8 (count=1, type=2 "image",
// data_size=1 (2B), tensor dims == tile dims, dim0 stride in elements).
// Issued once by one wave; completion tracked with TENSORcnt.
// ---------------------------------------------------------------------------
__device__ __forceinline__ void tdm_load_2d_to_lds(const void* gptr, unsigned lds_addr,
                                                   int tile_rows, int tile_cols,
                                                   int row_stride_elems) {
    unsigned long long ga = (unsigned long long)(size_t)gptr;
    u32x4_t g0;
    g0[0] = 1u;                                              // count=1, user desc
    g0[1] = lds_addr;                                        // lds_addr
    g0[2] = (unsigned)ga;                                    // global_addr[31:0]
    g0[3] = (unsigned)((ga >> 32) & 0x01FFFFFFull) | 0x80000000u; // [56:32] | type=2
    i32x8_t g1;
    g1[0] = 0x00010000;                        // wg_mask=0, data_size=1 (2 bytes)
    g1[1] = (int)((unsigned)tile_cols << 16);  // tensor_dim0[15:0] @ bits[63:48]
    g1[2] = (int)((unsigned)tile_rows << 16);  // tensor_dim1[15:0] @ bits[95:80]
    g1[3] = (int)((unsigned)tile_cols << 16);  // tile_dim0 @ bits[127:112]
    g1[4] = tile_rows;                         // tile_dim1 (tile_dim2 = 0)
    g1[5] = row_stride_elems;                  // tensor_dim0_stride[31:0]
    g1[6] = 0;
    g1[7] = 0;
    i32x4_t z4 = {0, 0, 0, 0};
#if defined(__clang_major__) && (__clang_major__ >= 23)
    i32x8_t z8 = {0, 0, 0, 0, 0, 0, 0, 0};
    __builtin_amdgcn_tensor_load_to_lds(g0, g1, z4, z4, z8, 0);
#else
    __builtin_amdgcn_tensor_load_to_lds(g0, g1, z4, z4, 0);
#endif
}

// ---------------------------------------------------------------------------
// Generic NT GEMM: C[M,N] (f32) = A[M,K](bf16) * Bt[N,K](bf16)^T  (+ C if acc)
// Block: 256 threads = 8 waves; block tile 128x128; wave tile 32x64.
// B tile (128 x 32, 8 KB) is TDM-loaded into double-buffered LDS and shared
// by all 8 waves; A fragments stream directly from global (L2-resident).
// grid = (N/128, M/128).  M,N,K multiples of (128,128,32) in this model.
// ---------------------------------------------------------------------------
__global__ void __launch_bounds__(256)
gemm_nt_kernel(const __bf16* __restrict__ A, int lda,
               const __bf16* __restrict__ Bt, int ldb,
               float* __restrict__ C, int ldc,
               int M, int N, int K, int accumulate) {
    (void)M; (void)N;
    __shared__ __align__(16) __bf16 Bs[2][128 * 32];   // 2 x 8 KB

    int tid  = threadIdx.x;
    int wave = tid >> 5, lane = tid & 31;
    int wm = wave >> 1;            // 0..3
    int wn = wave & 1;             // 0..1
    int row0 = blockIdx.y * 128 + wm * 32;
    int colb = blockIdx.x * 128;           // block column base

    v8f vzero = {0.f,0.f,0.f,0.f,0.f,0.f,0.f,0.f};
    v8f acc[2][4];
#pragma unroll
    for (int m = 0; m < 2; ++m)
#pragma unroll
        for (int n = 0; n < 4; ++n) acc[m][n] = vzero;

    const __bf16* arow0 = A + (size_t)row0 * lda;
    const __bf16* arow1 = A + (size_t)(row0 + 16) * lda;
    const __bf16* btile = Bt + (size_t)colb * ldb;

    int nkt = K / 32;

    // prologue: DMA tile 0
    if (wave == 0)
        tdm_load_2d_to_lds(btile, (unsigned)(size_t)&Bs[0][0], 128, 32, ldb);

    auto compute = [&](int kt) {
        const __bf16* bs = &Bs[kt & 1][(wn * 64) * 32];
        int k0 = kt * 32;
        v16bf a0 = load_frag(arow0 + k0, lda, lane);
        v16bf a1 = load_frag(arow1 + k0, lda, lane);
#pragma unroll
        for (int nf = 0; nf < 4; ++nf) {
            v16bf b = load_frag(bs + (size_t)(nf * 16) * 32, 32, lane);
            acc[0][nf] = wmma_bf16(a0, b, acc[0][nf]);
            acc[1][nf] = wmma_bf16(a1, b, acc[1][nf]);
        }
    };

    for (int kt = 0; kt < nkt - 1; ++kt) {
        if (wave == 0) {
            tdm_load_2d_to_lds(btile + (kt + 1) * 32,
                               (unsigned)(size_t)&Bs[(kt + 1) & 1][0], 128, 32, ldb);
            __builtin_amdgcn_s_wait_tensorcnt(1);   // tile kt complete
        }
        __syncthreads();
        compute(kt);
        __syncthreads();                            // Bs[kt&1] free for reuse
    }
    if (wave == 0) __builtin_amdgcn_s_wait_tensorcnt(0);
    __syncthreads();
    compute(nkt - 1);

    int rsub = (lane >> 4) * 8;
    int csub = lane & 15;
#pragma unroll
    for (int mf = 0; mf < 2; ++mf)
#pragma unroll
        for (int nf = 0; nf < 4; ++nf)
#pragma unroll
            for (int j = 0; j < 8; ++j) {
                int r = row0 + mf * 16 + rsub + j;
                int c = colb + wn * 64 + nf * 16 + csub;
                size_t idx = (size_t)r * ldc + c;
                float v = acc[mf][nf][j];
                if (accumulate) v += C[idx];
                C[idx] = v;
            }
}

// ---------------------------------------------------------------------------
// Fused RMS norm:  t = in (+bias) (+pre_add); y = rms(t)*w; opt silu; opt +res
// One block (256 threads) per row; cols <= 1024.
// ---------------------------------------------------------------------------
__global__ void fused_rms_kernel(const float* __restrict__ in,
                                 const float* __restrict__ bias,
                                 const float* __restrict__ pre_add,
                                 const float* __restrict__ w,
                                 const float* __restrict__ residual,
                                 float* __restrict__ out_f32,
                                 __bf16* __restrict__ out_bf16,
                                 int cols, int do_silu) {
    int row = blockIdx.x;
    int tid = threadIdx.x;
    __shared__ float tbuf[1024];
    __shared__ float red[8];
    const size_t base = (size_t)row * cols;

    float ss = 0.f;
    for (int c = tid; c < cols; c += blockDim.x) {
        float t = in[base + c];
        if (bias)    t += bias[c];
        if (pre_add) t += pre_add[base + c];
        tbuf[c] = t;
        ss += t * t;
    }
    for (int o = 16; o; o >>= 1) ss += __shfl_xor(ss, o, 32);
    if ((tid & 31) == 0) red[tid >> 5] = ss;
    __syncthreads();
    if (tid == 0) {
        float a = 0.f;
        int nw = blockDim.x >> 5;
        for (int i = 0; i < nw; ++i) a += red[i];
        red[0] = rsqrtf(a / (float)cols + EPSF);
    }
    __syncthreads();
    float scale = red[0];
    for (int c = tid; c < cols; c += blockDim.x) {
        float t = tbuf[c] * scale * w[c];
        if (do_silu)  t = t / (1.f + __expf(-t));
        if (residual) t += residual[base + c];
        if (out_f32)  out_f32[base + c]  = t;
        if (out_bf16) out_bf16[base + c] = (__bf16)t;
    }
}

// ---------------------------------------------------------------------------
// Small elementwise kernels
// ---------------------------------------------------------------------------
__global__ void cvt_bf16_kernel(const float* __restrict__ in, __bf16* __restrict__ out,
                                size_t n) {
    for (size_t i = blockIdx.x * (size_t)blockDim.x + threadIdx.x; i < n;
         i += (size_t)gridDim.x * blockDim.x)
        out[i] = (__bf16)in[i];
}

__global__ void transpose_cvt_kernel(const float* __restrict__ in, __bf16* __restrict__ out,
                                     int rows, int cols) {
    size_t total = (size_t)rows * cols;
    for (size_t i = blockIdx.x * (size_t)blockDim.x + threadIdx.x; i < total;
         i += (size_t)gridDim.x * blockDim.x) {
        size_t r = i / cols, c = i % cols;
        out[c * (size_t)rows + r] = (__bf16)in[i];
    }
}

__global__ void pack_pad_kernel(const float* __restrict__ xnorm, __bf16* __restrict__ xpad,
                                int Bn, int Sn, int Dn) {
    size_t total = (size_t)Bn * Sn * Dn;
    for (size_t i = blockIdx.x * (size_t)blockDim.x + threadIdx.x; i < total;
         i += (size_t)gridDim.x * blockDim.x) {
        size_t c = i % Dn;
        size_t s = (i / Dn) % Sn;
        size_t b = i / ((size_t)Dn * Sn);
        xpad[((b * (Sn + 2)) + s + 1) * Dn + c] = (__bf16)xnorm[i];
    }
}

__global__ void silu_bias_kernel(float* __restrict__ x, const float* __restrict__ bias,
                                 size_t rows, int cols) {
    size_t total = rows * (size_t)cols;
    for (size_t i = blockIdx.x * (size_t)blockDim.x + threadIdx.x; i < total;
         i += (size_t)gridDim.x * blockDim.x) {
        float v = x[i] + bias[i % cols];
        x[i] = v / (1.f + __expf(-v));
    }
}

__global__ void silu_mul_kernel(const float* __restrict__ h1, const float* __restrict__ h3,
                                __bf16* __restrict__ g, size_t n) {
    for (size_t i = blockIdx.x * (size_t)blockDim.x + threadIdx.x; i < n;
         i += (size_t)gridDim.x * blockDim.x) {
        float t = h1[i];
        g[i] = (__bf16)((t / (1.f + __expf(-t))) * h3[i]);
    }
}

// V (concat lc|gc per head) stored transposed: vt[(b*H+h)*64+d][s]  (bf16)
__global__ void v_prep_kernel(const float* __restrict__ qkv_lc,
                              const float* __restrict__ qkv_gc,
                              __bf16* __restrict__ vt,
                              int Bn, int Sn, int Hn) {
    size_t total = (size_t)Bn * Hn * 64 * Sn;
    for (size_t i = blockIdx.x * (size_t)blockDim.x + threadIdx.x; i < total;
         i += (size_t)gridDim.x * blockDim.x) {
        size_t s = i % Sn;
        size_t d = (i / Sn) % 64;
        size_t h = (i / ((size_t)Sn * 64)) % Hn;
        size_t b = i / ((size_t)Sn * 64 * Hn);
        size_t row = b * Sn + s;
        float val = (d < 32) ? qkv_lc[row * 1536 + 1024 + h * 32 + d]
                             : qkv_gc[row * 1536 + 1024 + h * 32 + (d - 32)];
        vt[i] = (__bf16)val;
    }
}

// ---------------------------------------------------------------------------
// Q/K prep: rms over 512 (lc & gc separately), RoPE, per-head concat + rms(64)
// One block (128 threads) per (b,s) row.  Output (B,H,S,64) bf16.
// ---------------------------------------------------------------------------
__global__ void __launch_bounds__(128)
qk_prep_kernel(const float* __restrict__ qkv_lc, const float* __restrict__ qkv_gc,
               int part_off,
               const float* __restrict__ w_lc, const float* __restrict__ w_gc,
               const float* __restrict__ w_head,
               const float* __restrict__ freqs,
               __bf16* __restrict__ out, int Sn, int Hn) {
    int row = blockIdx.x;             // b*S + s
    int s = row % Sn;
    int tid = threadIdx.x;
    __shared__ float v[1024];
    __shared__ float redbuf[8];
    __shared__ float scales[2];

    const size_t base = (size_t)row * 1536 + part_off;
    float ss_lc = 0.f, ss_gc = 0.f;
    for (int c = tid; c < 512; c += 128) {
        float a = qkv_lc[base + c];
        float b = qkv_gc[base + c];
        int h = c >> 5, d = c & 31;
        v[h * 64 + d]      = a;
        v[h * 64 + 32 + d] = b;
        ss_lc += a * a;
        ss_gc += b * b;
    }
    for (int o = 16; o; o >>= 1) {
        ss_lc += __shfl_xor(ss_lc, o, 32);
        ss_gc += __shfl_xor(ss_gc, o, 32);
    }
    int wave = tid >> 5, lane = tid & 31;
    if (lane == 0) { redbuf[wave] = ss_lc; redbuf[4 + wave] = ss_gc; }
    __syncthreads();
    if (tid == 0) {
        float a = 0.f, b = 0.f;
        for (int i = 0; i < 4; ++i) { a += redbuf[i]; b += redbuf[4 + i]; }
        scales[0] = rsqrtf(a / 512.f + EPSF);
        scales[1] = rsqrtf(b / 512.f + EPSF);
    }
    __syncthreads();
    float slc = scales[0], sgc = scales[1];

    // RoPE on 512 pairs (256 lc + 256 gc); pair i uses freqs[s,16] per head.
    for (int p = tid; p < 512; p += 128) {
        int part = p >> 8;            // 0 = lc, 1 = gc
        int q = p & 255;
        int h = q >> 4, i = q & 15;
        int bv = h * 64 + part * 32 + 2 * i;
        const float* wpart = part ? w_gc : w_lc;
        float sc = part ? sgc : slc;
        float x0 = v[bv]     * sc * wpart[h * 32 + 2 * i];
        float x1 = v[bv + 1] * sc * wpart[h * 32 + 2 * i + 1];
        float fr = freqs[(size_t)s * 16 + i];
        float cf = __cosf(fr), sf = __sinf(fr);
        v[bv]     = x0 * cf - x1 * sf;
        v[bv + 1] = x0 * sf + x1 * cf;
    }
    __syncthreads();

    // per-head rms over 64, write (B,H,S,64)
    int h = tid >> 3;
    int j0 = (tid & 7) * 8;
    float ssh = 0.f;
    for (int j = 0; j < 8; ++j) { float t = v[h * 64 + j0 + j]; ssh += t * t; }
    for (int o = 4; o; o >>= 1) ssh += __shfl_xor(ssh, o, 32);
    float sh = rsqrtf(ssh / 64.f + EPSF);
    int b = row / Sn;
    size_t obase = (((size_t)b * Hn + h) * Sn + s) * 64;
    for (int j = 0; j < 8; ++j) {
        int jj = j0 + j;
        out[obase + jj] = (__bf16)(v[h * 64 + jj] * sh * w_head[jj]);
    }
}

// ---------------------------------------------------------------------------
// Flash attention.  grid = (S/64, B*H), block = 128 (4 waves).
// xq/xk: (B,H,S,64) bf16;  vt: (B,H,64,S) bf16;  out: (B,S,H*64) bf16.
// ---------------------------------------------------------------------------
__global__ void __launch_bounds__(128)
attn_kernel(const __bf16* __restrict__ xq, const __bf16* __restrict__ xk,
            const __bf16* __restrict__ vt, __bf16* __restrict__ outp,
            int Sn, int Hn) {
    int qt = blockIdx.x;
    int bh = blockIdx.y;
    int tid = threadIdx.x;
    int wave = tid >> 5, lane = tid & 31;

    __shared__ __align__(16) float  Sb[64 * 64];
    __shared__ __align__(16) float  Ob[64 * 64];
    __shared__ __align__(16) __bf16 Pb[64 * 64];
    __shared__ float mrow[64];
    __shared__ float lrow[64];

    const __bf16* qbase  = xq + (((size_t)bh * Sn) + (size_t)qt * 64) * 64;
    const __bf16* kbase0 = xk + ((size_t)bh * Sn) * 64;
    const __bf16* vbase  = vt + ((size_t)bh * 64) * Sn;

    v16bf qa[2];
    qa[0] = load_frag(qbase + (size_t)(wave * 16) * 64 + 0, 64, lane);
    qa[1] = load_frag(qbase + (size_t)(wave * 16) * 64 + 32, 64, lane);

    for (int i = tid; i < 64 * 64; i += 128) Ob[i] = 0.f;
    if (tid < 64) { mrow[tid] = -1e30f; lrow[tid] = 0.f; }
    __syncthreads();

    int rsub = wave * 16 + ((lane >> 4) ? 8 : 0);
    int csub = lane & 15;
    v8f vzero = {0.f,0.f,0.f,0.f,0.f,0.f,0.f,0.f};

    for (int kt = 0; kt < Sn / 64; ++kt) {
        const __bf16* kb = kbase0 + (size_t)kt * 64 * 64;
        v8f acc[4] = {vzero, vzero, vzero, vzero};
#pragma unroll
        for (int ks = 0; ks < 2; ++ks) {
#pragma unroll
            for (int nf = 0; nf < 4; ++nf) {
                v16bf bfrag = load_frag(kb + (size_t)(nf * 16) * 64 + ks * 32, 64, lane);
                acc[nf] = wmma_bf16(qa[ks], bfrag, acc[nf]);
            }
        }
#pragma unroll
        for (int nf = 0; nf < 4; ++nf)
#pragma unroll
            for (int j = 0; j < 8; ++j)
                Sb[(rsub + j) * 64 + nf * 16 + csub] = acc[nf][j] * 0.125f;
        __syncthreads();

        if (tid < 64) {
            int r = tid;
            float mprev = mrow[r];
            float mx = mprev;
            for (int c = 0; c < 64; ++c) mx = fmaxf(mx, Sb[r * 64 + c]);
            float alpha = __expf(mprev - mx);
            float sum = 0.f;
            for (int c = 0; c < 64; ++c) {
                float p = __expf(Sb[r * 64 + c] - mx);
                Pb[r * 64 + c] = (__bf16)p;
                sum += p;
            }
            lrow[r] = lrow[r] * alpha + sum;
            mrow[r] = mx;
            for (int c = 0; c < 64; ++c) Ob[r * 64 + c] *= alpha;
        }
        __syncthreads();

        v8f pacc[4] = {vzero, vzero, vzero, vzero};
#pragma unroll
        for (int ks = 0; ks < 2; ++ks) {
            v16bf pa = load_frag(&Pb[(wave * 16) * 64 + ks * 32], 64, lane);
#pragma unroll
            for (int nf = 0; nf < 4; ++nf) {
                v16bf vb = load_frag(vbase + (size_t)(nf * 16) * Sn + (size_t)kt * 64 + ks * 32,
                                     Sn, lane);
                pacc[nf] = wmma_bf16(pa, vb, pacc[nf]);
            }
        }
#pragma unroll
        for (int nf = 0; nf < 4; ++nf)
#pragma unroll
            for (int j = 0; j < 8; ++j)
                Ob[(rsub + j) * 64 + nf * 16 + csub] += pacc[nf][j];
        __syncthreads();
    }

    int b = bh / Hn, h = bh % Hn;
    for (int i = tid; i < 64 * 64; i += 128) {
        int r = i >> 6, c = i & 63;
        float o = Ob[i] / lrow[r];
        outp[((size_t)(b * Sn + qt * 64 + r)) * (Hn * 64) + h * 64 + c] = (__bf16)o;
    }
}

// ---------------------------------------------------------------------------
// Host orchestration
// ---------------------------------------------------------------------------
extern "C" void kernel_launch(void* const* d_in, const int* in_sizes, int n_in,
                              void* d_out, int out_size, void* d_ws, size_t ws_size,
                              hipStream_t stream) {
    (void)in_sizes; (void)n_in; (void)out_size; (void)ws_size;

    enum { Bc = 2, Sc = 2048, Dc = 1024, Hc = 16, INTERc = 1024, QDc = 512, HIDc = 4096 };
    const size_t Rr = (size_t)Bc * Sc;     // 4096 rows

    const float* x          = (const float*)d_in[0];
    const float* freqs      = (const float*)d_in[1];
    const float* w_pre      = (const float*)d_in[2];
    const float* lc_conv_w  = (const float*)d_in[3];
    const float* lc_conv_b  = (const float*)d_in[4];
    const float* gc_w       = (const float*)d_in[5];
    const float* gc_b       = (const float*)d_in[6];
    const float* w_lc_norm  = (const float*)d_in[7];
    const float* w_gc_norm  = (const float*)d_in[8];
    const float* attn_lc_w  = (const float*)d_in[9];
    const float* attn_gc_w  = (const float*)d_in[10];
    const float* w_q_lc     = (const float*)d_in[11];
    const float* w_k_lc     = (const float*)d_in[12];
    const float* w_q_gc     = (const float*)d_in[13];
    const float* w_k_gc     = (const float*)d_in[14];
    const float* w_q_head   = (const float*)d_in[15];
    const float* w_k_head   = (const float*)d_in[16];
    const float* local_out_w  = (const float*)d_in[17];
    const float* local_out_b  = (const float*)d_in[18];
    const float* w_lc_out   = (const float*)d_in[19];
    const float* global_out_w = (const float*)d_in[20];
    const float* global_out_b = (const float*)d_in[21];
    const float* w_gc_out   = (const float*)d_in[22];
    const float* w_lc_ffn   = (const float*)d_in[23];
    const float* w_gc_ffn   = (const float*)d_in[24];
    const float* ff_lc_w1   = (const float*)d_in[25];
    const float* ff_lc_w2   = (const float*)d_in[26];
    const float* ff_lc_w3   = (const float*)d_in[27];
    const float* ff_gc_w1   = (const float*)d_in[28];
    const float* ff_gc_w2   = (const float*)d_in[29];
    const float* ff_gc_w3   = (const float*)d_in[30];
    const float* fin_w      = (const float*)d_in[31];
    const float* fin_b      = (const float*)d_in[32];
    const float* w_fin_norm = (const float*)d_in[33];

    char* ws = (char*)d_ws;
    size_t cur = 0;
    auto alloc = [&](size_t bytes) -> void* {
        void* p = ws + cur;
        cur = (cur + bytes + 255) & ~(size_t)255;
        return p;
    };
    auto BF = [](size_t n) { return n * sizeof(__bf16); };
    auto F4 = [](size_t n) { return n * sizeof(float); };

    // bf16 weights
    __bf16* wT_conv   = (__bf16*)alloc(BF((size_t)3 * INTERc * Dc));
    __bf16* wb_gc     = (__bf16*)alloc(BF((size_t)INTERc * Dc));
    __bf16* wb_alc    = (__bf16*)alloc(BF((size_t)1536 * INTERc));
    __bf16* wb_agc    = (__bf16*)alloc(BF((size_t)1536 * INTERc));
    __bf16* wb_lout   = (__bf16*)alloc(BF((size_t)INTERc * 1024));
    __bf16* wb_gout   = (__bf16*)alloc(BF((size_t)INTERc * 1024));
    __bf16* wb_fl1    = (__bf16*)alloc(BF((size_t)HIDc * Dc));
    __bf16* wb_fl2    = (__bf16*)alloc(BF((size_t)Dc * HIDc));
    __bf16* wb_fl3    = (__bf16*)alloc(BF((size_t)HIDc * Dc));
    __bf16* wb_fg1    = (__bf16*)alloc(BF((size_t)HIDc * Dc));
    __bf16* wb_fg2    = (__bf16*)alloc(BF((size_t)Dc * HIDc));
    __bf16* wb_fg3    = (__bf16*)alloc(BF((size_t)HIDc * Dc));
    __bf16* wb_fin    = (__bf16*)alloc(BF((size_t)Dc * 2 * INTERc));
    // activations
    float*  xnorm     = (float*)alloc(F4(Rr * Dc));
    __bf16* xpad      = (__bf16*)alloc(BF((size_t)Bc * (Sc + 2) * Dc));
    float*  local_c   = (float*)alloc(F4(Rr * INTERc));
    float*  global_c  = (float*)alloc(F4(Rr * INTERc));
    float*  lc_n      = (float*)alloc(F4(Rr * INTERc));
    float*  gc_n      = (float*)alloc(F4(Rr * INTERc));
    __bf16* lcn_bf    = (__bf16*)alloc(BF(Rr * INTERc));
    __bf16* gcn_bf    = (__bf16*)alloc(BF(Rr * INTERc));
    float*  qkv_lc    = (float*)alloc(F4(Rr * 1536));
    float*  qkv_gc    = (float*)alloc(F4(Rr * 1536));
    __bf16* xq        = (__bf16*)alloc(BF(Rr * 1024));
    __bf16* xk        = (__bf16*)alloc(BF(Rr * 1024));
    __bf16* vtb       = (__bf16*)alloc(BF(Rr * 1024));
    __bf16* attnout   = (__bf16*)alloc(BF(Rr * 1024));
    float*  proj      = (float*)alloc(F4(Rr * 1024));
    float*  xbr       = (float*)alloc(F4(Rr * 1024));
    __bf16* t_lc      = (__bf16*)alloc(BF(Rr * 1024));
    __bf16* t_gc      = (__bf16*)alloc(BF(Rr * 1024));
    float*  h1        = (float*)alloc(F4(Rr * HIDc));
    float*  h3        = (float*)alloc(F4(Rr * HIDc));
    __bf16* gbuf      = (__bf16*)alloc(BF(Rr * HIDc));
    __bf16* lc_bf     = (__bf16*)alloc(BF(Rr * INTERc));
    __bf16* gc_bf     = (__bf16*)alloc(BF(Rr * INTERc));
    float*  finC      = (float*)alloc(F4(Rr * Dc));

    const int EB = 4096, ET = 256;   // elementwise grid

    // --- weight conversion / transposition ---------------------------------
    for (int t = 0; t < 3; ++t)
        transpose_cvt_kernel<<<EB, ET, 0, stream>>>(lc_conv_w + (size_t)t * Dc * INTERc,
                                                    wT_conv + (size_t)t * INTERc * Dc,
                                                    Dc, INTERc);
    cvt_bf16_kernel<<<EB, ET, 0, stream>>>(gc_w,  wb_gc,  (size_t)INTERc * Dc);
    cvt_bf16_kernel<<<EB, ET, 0, stream>>>(attn_lc_w, wb_alc, (size_t)1536 * INTERc);
    cvt_bf16_kernel<<<EB, ET, 0, stream>>>(attn_gc_w, wb_agc, (size_t)1536 * INTERc);
    cvt_bf16_kernel<<<EB, ET, 0, stream>>>(local_out_w,  wb_lout, (size_t)INTERc * 1024);
    cvt_bf16_kernel<<<EB, ET, 0, stream>>>(global_out_w, wb_gout, (size_t)INTERc * 1024);
    cvt_bf16_kernel<<<EB, ET, 0, stream>>>(ff_lc_w1, wb_fl1, (size_t)HIDc * Dc);
    cvt_bf16_kernel<<<EB, ET, 0, stream>>>(ff_lc_w2, wb_fl2, (size_t)Dc * HIDc);
    cvt_bf16_kernel<<<EB, ET, 0, stream>>>(ff_lc_w3, wb_fl3, (size_t)HIDc * Dc);
    cvt_bf16_kernel<<<EB, ET, 0, stream>>>(ff_gc_w1, wb_fg1, (size_t)HIDc * Dc);
    cvt_bf16_kernel<<<EB, ET, 0, stream>>>(ff_gc_w2, wb_fg2, (size_t)Dc * HIDc);
    cvt_bf16_kernel<<<EB, ET, 0, stream>>>(ff_gc_w3, wb_fg3, (size_t)HIDc * Dc);
    cvt_bf16_kernel<<<EB, ET, 0, stream>>>(fin_w,   wb_fin, (size_t)Dc * 2 * INTERc);

    // --- pre-norm + padded bf16 copy ---------------------------------------
    fused_rms_kernel<<<Rr, 256, 0, stream>>>(x, nullptr, nullptr, w_pre, nullptr,
                                             xnorm, nullptr, Dc, 0);
    hipMemsetAsync(xpad, 0, BF((size_t)Bc * (Sc + 2) * Dc), stream);
    pack_pad_kernel<<<EB, ET, 0, stream>>>(xnorm, xpad, Bc, Sc, Dc);

    // --- conv1d(SAME,3) as 3 accumulating GEMMs per batch ------------------
    for (int b = 0; b < Bc; ++b)
        for (int t = 0; t < 3; ++t)
            gemm_nt_kernel<<<dim3(INTERc / 128, Sc / 128), 256, 0, stream>>>(
                xpad + ((size_t)b * (Sc + 2) + t) * Dc, Dc,
                wT_conv + (size_t)t * INTERc * Dc, Dc,
                local_c + (size_t)b * Sc * INTERc, INTERc,
                Sc, INTERc, Dc, t > 0 ? 1 : 0);

    // --- global channel GEMM ------------------------------------------------
    for (int b = 0; b < Bc; ++b)
        gemm_nt_kernel<<<dim3(INTERc / 128, Sc / 128), 256, 0, stream>>>(
            xpad + ((size_t)b * (Sc + 2) + 1) * Dc, Dc,
            wb_gc, Dc,
            global_c + (size_t)b * Sc * INTERc, INTERc,
            Sc, INTERc, Dc, 0);

    silu_bias_kernel<<<EB, ET, 0, stream>>>(local_c,  lc_conv_b, Rr, INTERc);
    silu_bias_kernel<<<EB, ET, 0, stream>>>(global_c, gc_b,      Rr, INTERc);

    fused_rms_kernel<<<Rr, 256, 0, stream>>>(local_c,  nullptr, nullptr, w_lc_norm,
                                             nullptr, lc_n, lcn_bf, INTERc, 0);
    fused_rms_kernel<<<Rr, 256, 0, stream>>>(global_c, nullptr, nullptr, w_gc_norm,
                                             nullptr, gc_n, gcn_bf, INTERc, 0);

    // --- QKV projections ----------------------------------------------------
    gemm_nt_kernel<<<dim3(1536 / 128, Rr / 128), 256, 0, stream>>>(
        lcn_bf, INTERc, wb_alc, INTERc, qkv_lc, 1536, (int)Rr, 1536, INTERc, 0);
    gemm_nt_kernel<<<dim3(1536 / 128, Rr / 128), 256, 0, stream>>>(
        gcn_bf, INTERc, wb_agc, INTERc, qkv_gc, 1536, (int)Rr, 1536, INTERc, 0);

    // --- Q/K prep (rms + rope + head rms), V transpose ----------------------
    qk_prep_kernel<<<Rr, 128, 0, stream>>>(qkv_lc, qkv_gc, 0,   w_q_lc, w_q_gc,
                                           w_q_head, freqs, xq, Sc, Hc);
    qk_prep_kernel<<<Rr, 128, 0, stream>>>(qkv_lc, qkv_gc, QDc, w_k_lc, w_k_gc,
                                           w_k_head, freqs, xk, Sc, Hc);
    v_prep_kernel<<<EB, ET, 0, stream>>>(qkv_lc, qkv_gc, vtb, Bc, Sc, Hc);

    // --- attention ----------------------------------------------------------
    attn_kernel<<<dim3(Sc / 64, Bc * Hc), 128, 0, stream>>>(xq, xk, vtb, attnout, Sc, Hc);

    // --- local branch: out-proj, norms, FFN ---------------------------------
    gemm_nt_kernel<<<dim3(1024 / 128, Rr / 128), 256, 0, stream>>>(
        attnout, 1024, wb_lout, 1024, proj, 1024, (int)Rr, 1024, 1024, 0);
    fused_rms_kernel<<<Rr, 256, 0, stream>>>(proj, local_out_b, nullptr, w_lc_out,
                                             nullptr, xbr, nullptr, 1024, 0);
    fused_rms_kernel<<<Rr, 256, 0, stream>>>(xbr, nullptr, lc_n, w_lc_ffn,
                                             nullptr, nullptr, t_lc, 1024, 0);
    gemm_nt_kernel<<<dim3(HIDc / 128, Rr / 128), 256, 0, stream>>>(
        t_lc, 1024, wb_fl1, Dc, h1, HIDc, (int)Rr, HIDc, Dc, 0);
    gemm_nt_kernel<<<dim3(HIDc / 128, Rr / 128), 256, 0, stream>>>(
        t_lc, 1024, wb_fl3, Dc, h3, HIDc, (int)Rr, HIDc, Dc, 0);
    silu_mul_kernel<<<EB, ET, 0, stream>>>(h1, h3, gbuf, Rr * HIDc);
    gemm_nt_kernel<<<dim3(INTERc / 128, Rr / 128), 256, 0, stream>>>(
        gbuf, HIDc, wb_fl2, HIDc, local_c, INTERc, (int)Rr, INTERc, HIDc, 1);

    // --- global branch ------------------------------------------------------
    gemm_nt_kernel<<<dim3(1024 / 128, Rr / 128), 256, 0, stream>>>(
        attnout, 1024, wb_gout, 1024, proj, 1024, (int)Rr, 1024, 1024, 0);
    fused_rms_kernel<<<Rr, 256, 0, stream>>>(proj, global_out_b, nullptr, w_gc_out,
                                             nullptr, xbr, nullptr, 1024, 0);
    fused_rms_kernel<<<Rr, 256, 0, stream>>>(xbr, nullptr, gc_n, w_gc_ffn,
                                             nullptr, nullptr, t_gc, 1024, 0);
    gemm_nt_kernel<<<dim3(HIDc / 128, Rr / 128), 256, 0, stream>>>(
        t_gc, 1024, wb_fg1, Dc, h1, HIDc, (int)Rr, HIDc, Dc, 0);
    gemm_nt_kernel<<<dim3(HIDc / 128, Rr / 128), 256, 0, stream>>>(
        t_gc, 1024, wb_fg3, Dc, h3, HIDc, (int)Rr, HIDc, Dc, 0);
    silu_mul_kernel<<<EB, ET, 0, stream>>>(h1, h3, gbuf, Rr * HIDc);
    gemm_nt_kernel<<<dim3(INTERc / 128, Rr / 128), 256, 0, stream>>>(
        gbuf, HIDc, wb_fg2, HIDc, global_c, INTERc, (int)Rr, INTERc, HIDc, 1);

    // --- final: concat is split into two K=1024 GEMMs into the same C -------
    cvt_bf16_kernel<<<EB, ET, 0, stream>>>(local_c,  lc_bf, Rr * INTERc);
    cvt_bf16_kernel<<<EB, ET, 0, stream>>>(global_c, gc_bf, Rr * INTERc);
    gemm_nt_kernel<<<dim3(Dc / 128, Rr / 128), 256, 0, stream>>>(
        lc_bf, INTERc, wb_fin, 2 * INTERc, finC, Dc, (int)Rr, Dc, INTERc, 0);
    gemm_nt_kernel<<<dim3(Dc / 128, Rr / 128), 256, 0, stream>>>(
        gc_bf, INTERc, wb_fin + INTERc, 2 * INTERc, finC, Dc, (int)Rr, Dc, INTERc, 1);

    fused_rms_kernel<<<Rr, 256, 0, stream>>>(finC, fin_b, nullptr, w_fin_norm,
                                             x, (float*)d_out, nullptr, Dc, 1);
}
